// GSNet_75402445849113
// MI455X (gfx1250) — matrix operand order repaired and checked by reading, not compiled, not run on hardware
//
#include <hip/hip_runtime.h>
#include <hip/hip_bf16.h>

// ---------------------------------------------------------------------------
// GSNet forward for MI455X (gfx1250, wave32, WMMA).
//
// Live dataflow only (the hk/memory branch never reaches the output):
//   nv1/nv2 -> Performer features qp/kp (batch-independent, N=50000)
//   xc = [xt@w_in^T, node_emb]; v = relu(xc@fc_w^T)
//   kv = kp^T @ v (per batch, K=N GEMM); ksum = sum_n kp
//   h  = (qp@kv)/(qp.ksum) + xc; LN; out = relu([xc,h]) @ w_reg^T + b_reg
//
// ~12 GFLOP vs ~150 MB traffic -> memory-bound at 23.3 TB/s; all K>=32 GEMMs
// use v_wmma_f32_16x16x32_f16 (f16 in / f32 accum); exp/LN/div stay f32 VALU.
// All cross-block reductions use partial buffers + fixed-order reduce kernels
// so replays are bit-deterministic (no float atomics).
// ---------------------------------------------------------------------------

#define NN    50000
#define NPAD  50016            // N rounded up to 32 (WMMA K-chunk)
#define NT    3125             // NN / 16 node tiles
#define BB    4
#define HH    32
#define H2    64
#define RR    64
#define TT    12
#define SEG   64               // K-segments for the kv GEMM
#define CPS   25               // chunks (of 32 nodes) per segment
#define CHUNKS (NPAD/32)       // 1563

typedef __attribute__((ext_vector_type(16))) _Float16 v16h;
typedef __attribute__((ext_vector_type(8)))  float    v8f;

// --- WMMA operand index helpers (CDNA5 ISA 7.12.2 layouts, wave32) ---------
__device__ __forceinline__ int kidxA(int i, int lhi) {   // A 16x32 f16
  return 2*i + (i >= 4 ? 8 : 0) + (lhi ? 8 : 0);
}
__device__ __forceinline__ int kidxB(int i, int lhi) {   // B 32x16 f16
  return 2*i + (i >= 4 ? 8 : 0) + (lhi ? 8 : 0);
}

__device__ __forceinline__ v8f wmma32(v16h a, v16h b, v8f c) {
  return __builtin_amdgcn_wmma_f32_16x16x32_f16(false, a, false, b,
                                                (short)0, c, false, false);
}

// A fragment from row-major f32: element A[row][k0+kidxA], zero-pad past klim
__device__ __forceinline__ v16h fragA_f32(const float* p, int stride, int row,
                                          int lhi, int k0, float scale, int klim) {
  v16h v = {};
  #pragma unroll
  for (int i = 0; i < 8; ++i) {
    int k = k0 + kidxA(i, lhi);
    float a = (k     < klim) ? p[(size_t)row*stride + k    ] * scale : 0.f;
    float b = (k + 1 < klim) ? p[(size_t)row*stride + k + 1] * scale : 0.f;
    v[2*i] = (_Float16)a;  v[2*i+1] = (_Float16)b;
  }
  return v;
}
// B fragment where B[k][n] = W[n][k], W row-major (i.e. x @ W^T GEMMs)
__device__ __forceinline__ v16h fragBT_f32(const float* p, int stride, int col,
                                           int lhi, int k0, int klim) {
  v16h v = {};
  #pragma unroll
  for (int i = 0; i < 8; ++i) {
    int k = k0 + kidxB(i, lhi);
    float a = (k     < klim) ? p[(size_t)col*stride + k    ] : 0.f;
    float b = (k + 1 < klim) ? p[(size_t)col*stride + k + 1] : 0.f;
    v[2*i] = (_Float16)a;  v[2*i+1] = (_Float16)b;
  }
  return v;
}
// B fragment from k-major f32 matrix B[k][col] (used for kv)
__device__ __forceinline__ v16h fragB_f32(const float* p, int stride, int col,
                                          int lhi, int k0) {
  v16h v = {};
  #pragma unroll
  for (int i = 0; i < 8; ++i) {
    int k = k0 + kidxB(i, lhi);
    v[2*i]   = (_Float16)p[(size_t)k      *stride + col];
    v[2*i+1] = (_Float16)p[(size_t)(k + 1)*stride + col];
  }
  return v;
}
__device__ __forceinline__ v16h fragA_f16(const _Float16* p, int stride, int row,
                                          int lhi, int k0) {
  v16h v = {};
  #pragma unroll
  for (int i = 0; i < 8; ++i) {
    int k = k0 + kidxA(i, lhi);
    v[2*i]   = p[(size_t)row*stride + k];
    v[2*i+1] = p[(size_t)row*stride + k + 1];
  }
  return v;
}
__device__ __forceinline__ v16h fragBT_f16(const _Float16* p, int stride, int col,
                                           int lhi, int k0) {
  v16h v = {};
  #pragma unroll
  for (int i = 0; i < 8; ++i) {
    int k = k0 + kidxB(i, lhi);
    v[2*i]   = p[(size_t)col*stride + k];
    v[2*i+1] = p[(size_t)col*stride + k + 1];
  }
  return v;
}

// ---------------------------------------------------------------------------
// k0: zero the node-padding region of v^T (nodes NN..NPAD-1)
__global__ void k0_zero(_Float16* __restrict__ hpT) {
  const int idx = blockIdx.x * 256 + threadIdx.x;          // 4096 total
  if (idx >= BB * H2 * (NPAD - NN)) return;
  const int bc = idx / (NPAD - NN);
  const int n  = NN + idx % (NPAD - NN);
  hpT[(size_t)bc * NPAD + n] = (_Float16)0.f;
}

// ---------------------------------------------------------------------------
// k1: per 16-node tile (batch-independent): nv1/nv2 -> dd -> qp, e_k, blockmax
__global__ void k1_node(const float* __restrict__ node_emb,
                        const float* __restrict__ w1, const float* __restrict__ b1,
                        const float* __restrict__ w2, const float* __restrict__ b2,
                        const float* __restrict__ rm1,
                        _Float16* __restrict__ qp, float* __restrict__ ek,
                        float* __restrict__ bmax) {
  __shared__ float s_nv[16][HH];
  __shared__ float s_dd[16][RR];
  __shared__ float s_red[32];
  const int lane = threadIdx.x;
  const int lm = lane & 15, lhi = lane >> 4;
  const int n0 = blockIdx.x * 16;
  const int node = lane >> 1, half = lane & 1;
  const float nrm  = 0.42044820762685725f;   // 32^-0.25
  const float nrm2 = nrm * nrm;

  v16h a_node = fragA_f32(node_emb, HH, n0 + lm, lhi, 0, 1.0f, HH);

  #pragma unroll 1
  for (int pass = 0; pass < 2; ++pass) {
    const float* wmat = pass ? w2 : w1;
    const float* bias = pass ? b2 : b1;
    // nv = node_emb @ w^T + b  (16x32 tile, 2 col tiles)
    #pragma unroll
    for (int t = 0; t < 2; ++t) {
      v16h bf = fragBT_f32(wmat, HH, lm + 16*t, lhi, 0, HH);
      v8f d = {0.f,0.f,0.f,0.f,0.f,0.f,0.f,0.f};
      d = wmma32(a_node, bf, d);
      #pragma unroll
      for (int v = 0; v < 8; ++v)
        s_nv[v + 8*lhi][lm + 16*t] = d[v] + bias[lm + 16*t];
    }
    __syncthreads();
    // dd = (nv * nrm) @ rm1^T  (16x64, 4 col tiles)
    v16h a2 = fragA_f32(&s_nv[0][0], HH, lm, lhi, 0, nrm, HH);
    #pragma unroll
    for (int t = 0; t < 4; ++t) {
      v16h bf = fragBT_f32(rm1, HH, lm + 16*t, lhi, 0, HH);
      v8f d = {0.f,0.f,0.f,0.f,0.f,0.f,0.f,0.f};
      d = wmma32(a2, bf, d);
      #pragma unroll
      for (int v = 0; v < 8; ++v)
        s_dd[v + 8*lhi][lm + 16*t] = d[v];
    }
    __syncthreads();
    // diag = 0.5*nrm^2*sum(nv^2), per node (2 lanes/node)
    float sq = 0.f;
    for (int k = half*16; k < half*16 + 16; ++k) { float xx = s_nv[node][k]; sq += xx*xx; }
    s_red[lane] = sq; __syncthreads();
    const float diag = 0.5f * nrm2 * (s_red[2*node] + s_red[2*node + 1]);
    __syncthreads();
    float mx = -1e30f;
    for (int c = half*32; c < half*32 + 32; ++c) mx = fmaxf(mx, s_dd[node][c]);
    s_red[lane] = mx; __syncthreads();
    if (pass == 0) {
      const float rowm = fmaxf(s_red[2*node], s_red[2*node + 1]);   // per-query max
      for (int c = half*32; c < half*32 + 32; ++c) {
        float v = 0.125f * (expf(s_dd[node][c] - diag - rowm) + 1e-4f);
        qp[(size_t)(n0 + node)*RR + c] = (_Float16)v;
      }
    } else {
      for (int c = half*32; c < half*32 + 32; ++c)
        ek[(size_t)(n0 + node)*RR + c] = s_dd[node][c] - diag;      // stabilized later
      if (lane == 0) {
        float m = -1e30f;
        for (int i = 0; i < 32; ++i) m = fmaxf(m, s_red[i]);
        bmax[blockIdx.x] = m;                                       // max of raw dd_k
      }
    }
    __syncthreads();
  }
}

// ---------------------------------------------------------------------------
// k2: global key stabilizer M = max over all dd_k
__global__ void k2_max(const float* __restrict__ bmax, float* __restrict__ Mg) {
  __shared__ float s[256];
  float m = -1e30f;
  for (int i = threadIdx.x; i < NT; i += 256) m = fmaxf(m, bmax[i]);
  s[threadIdx.x] = m; __syncthreads();
  for (int st = 128; st > 0; st >>= 1) {
    if (threadIdx.x < st) s[threadIdx.x] = fmaxf(s[threadIdx.x], s[threadIdx.x + st]);
    __syncthreads();
  }
  if (threadIdx.x == 0) Mg[0] = s[0];
}

// ---------------------------------------------------------------------------
// k3: kp^T[r][n] = ratio*(exp(e_k - M)+eps), zero node padding
__global__ void k3_kp(const float* __restrict__ ek, const float* __restrict__ Mg,
                      _Float16* __restrict__ kpT) {
  const size_t idx = (size_t)blockIdx.x * 256 + threadIdx.x;
  if (idx >= (size_t)RR * NPAD) return;
  const int r = (int)(idx / NPAD);
  const int n = (int)(idx % NPAD);
  float v = 0.f;
  if (n < NN) v = 0.125f * (expf(ek[(size_t)n*RR + r] - Mg[0]) + 1e-4f);
  kpT[idx] = (_Float16)v;
}

// k3b: ksum[r] = sum_n kp[n][r]   (deterministic in-block tree)
__global__ void k3b_ksum(const _Float16* __restrict__ kpT, float* __restrict__ ksum) {
  __shared__ float s[256];
  const int r = blockIdx.x;
  float acc = 0.f;
  for (int n = threadIdx.x; n < NPAD; n += 256)
    acc += (float)kpT[(size_t)r*NPAD + n];
  s[threadIdx.x] = acc; __syncthreads();
  for (int st = 128; st > 0; st >>= 1) {
    if (threadIdx.x < st) s[threadIdx.x] += s[threadIdx.x + st];
    __syncthreads();
  }
  if (threadIdx.x == 0) ksum[r] = s[0];
}

// ---------------------------------------------------------------------------
// k4: per (b, 16-node tile): xc = [xt@w_in^T+b, node_emb]; v = relu(xc@fc_w^T+b)
//     store v transposed as f16: hpT[b][c][n] (n contiguous for the kv GEMM)
__global__ void k4_xc(const float* __restrict__ x, const float* __restrict__ node_emb,
                      const float* __restrict__ w_input, const float* __restrict__ b_input,
                      const float* __restrict__ fc_w, const float* __restrict__ fc_b,
                      _Float16* __restrict__ hpT) {
  __shared__ float s_xc[16][H2];
  const int lane = threadIdx.x, lm = lane & 15, lhi = lane >> 4;
  const int b = blockIdx.x / NT, n0 = (blockIdx.x % NT) * 16;

  v16h ax = {};                                  // xt tile, K padded 12 -> 32
  #pragma unroll
  for (int i = 0; i < 8; ++i) {
    int k = kidxA(i, lhi);
    size_t base = ((size_t)b*NN + (n0 + lm)) * TT;
    float v0 = (k     < TT) ? x[(base + k    ) * 3] : 0.f;   // x[b][n][t][0]
    float v1 = (k + 1 < TT) ? x[(base + k + 1) * 3] : 0.f;
    ax[2*i] = (_Float16)v0;  ax[2*i+1] = (_Float16)v1;
  }
  #pragma unroll
  for (int t = 0; t < 2; ++t) {
    v16h bf = fragBT_f32(w_input, TT, lm + 16*t, lhi, 0, TT);
    v8f d = {0.f,0.f,0.f,0.f,0.f,0.f,0.f,0.f};
    d = wmma32(ax, bf, d);
    #pragma unroll
    for (int v = 0; v < 8; ++v)
      s_xc[v + 8*lhi][lm + 16*t] = d[v] + b_input[lm + 16*t];
  }
  {
    const int node = lane >> 1, half = lane & 1;
    for (int k = half*16; k < half*16 + 16; ++k)
      s_xc[node][HH + k] = node_emb[(size_t)(n0 + node)*HH + k];
  }
  __syncthreads();
  v16h a0 = fragA_f32(&s_xc[0][0], H2, lm, lhi, 0,  1.f, H2);
  v16h a1 = fragA_f32(&s_xc[0][0], H2, lm, lhi, 32, 1.f, H2);
  #pragma unroll
  for (int t = 0; t < 4; ++t) {
    v8f d = {0.f,0.f,0.f,0.f,0.f,0.f,0.f,0.f};
    d = wmma32(a0, fragBT_f32(fc_w, H2, lm + 16*t, lhi, 0,  H2), d);
    d = wmma32(a1, fragBT_f32(fc_w, H2, lm + 16*t, lhi, 32, H2), d);
    const int c = lm + 16*t;
    #pragma unroll
    for (int v = 0; v < 8; ++v) {
      float h = fmaxf(d[v] + fc_b[c], 0.f);
      hpT[((size_t)b*H2 + c)*NPAD + n0 + v + 8*lhi] = (_Float16)h;
    }
  }
}

// ---------------------------------------------------------------------------
// k5: kv partials: C(64x64) += kp^T(64xK) @ v(Kx64) over one K-segment
__global__ void k5_kv(const _Float16* __restrict__ kpT, const _Float16* __restrict__ hpT,
                      float* __restrict__ kvpart) {
  const int lane = threadIdx.x, lm = lane & 15, lhi = lane >> 4;
  int id = blockIdx.x;
  const int seg  = id % SEG;  id /= SEG;
  const int tile = id % 16;   const int b = id / 16;
  const int r0 = (tile >> 2) * 16, c0 = (tile & 3) * 16;
  int cs = seg * CPS, ce = cs + CPS;  if (ce > CHUNKS) ce = CHUNKS;
  const _Float16* hb = hpT + (size_t)b * H2 * NPAD;
  v8f acc = {0.f,0.f,0.f,0.f,0.f,0.f,0.f,0.f};
  for (int ch = cs; ch < ce; ++ch) {
    const int k0 = ch * 32;
    v16h a  = fragA_f16 (kpT, NPAD, r0 + lm, lhi, k0);
    v16h bf = fragBT_f16(hb,  NPAD, c0 + lm, lhi, k0);   // v^T rows are cols of B
    acc = wmma32(a, bf, acc);
  }
  float* dst = kvpart + (((size_t)(b*16 + tile))*SEG + seg) * 256;
  #pragma unroll
  for (int v = 0; v < 8; ++v) dst[v*32 + lane] = acc[v];
}

// k5b: deterministic segment reduce -> kv[b][r][c]
__global__ void k5b_red(const float* __restrict__ kvpart, float* __restrict__ kv) {
  const int idx = blockIdx.x * 256 + threadIdx.x;   // 16384 total
  const int b = idx >> 12;
  const int t = (idx >> 8) & 15;
  const int e = idx & 255;
  const float* src = kvpart + ((size_t)(b*16 + t))*SEG*256 + e;
  float s = 0.f;
  for (int sg = 0; sg < SEG; ++sg) s += src[(size_t)sg * 256];
  const int v = e >> 5, lane = e & 31, lm = lane & 15, lhi = lane >> 4;
  const int r = (t >> 2)*16 + v + 8*lhi;
  const int c = (t & 3)*16 + lm;
  kv[((size_t)b*H2 + r)*H2 + c] = s;
}

// ---------------------------------------------------------------------------
// k6: per (b, tile): recompute xc; num = qp@kv; h = num/den + xc; LN;
//     out = relu([xc,h]) @ w_reg^T + b_reg  (12 output cols, rows 12..15 zeroed)
__global__ void k6_final(const float* __restrict__ x, const float* __restrict__ node_emb,
                         const float* __restrict__ w_input, const float* __restrict__ b_input,
                         const _Float16* __restrict__ qp, const float* __restrict__ kv,
                         const float* __restrict__ ksum,
                         const float* __restrict__ ln_g, const float* __restrict__ ln_b,
                         const float* __restrict__ w_reg, const float* __restrict__ b_reg,
                         float* __restrict__ out) {
  __shared__ float s_xc[16][H2];
  __shared__ float s_h[16][H2];
  __shared__ float s_red[32];
  const int lane = threadIdx.x, lm = lane & 15, lhi = lane >> 4;
  const int b = blockIdx.x / NT, n0 = (blockIdx.x % NT) * 16;
  const int node = lane >> 1, half = lane & 1;

  // ---- recompute xc (cheaper than storing 51 MB of it) ----
  v16h ax = {};
  #pragma unroll
  for (int i = 0; i < 8; ++i) {
    int k = kidxA(i, lhi);
    size_t base = ((size_t)b*NN + (n0 + lm)) * TT;
    float v0 = (k     < TT) ? x[(base + k    ) * 3] : 0.f;
    float v1 = (k + 1 < TT) ? x[(base + k + 1) * 3] : 0.f;
    ax[2*i] = (_Float16)v0;  ax[2*i+1] = (_Float16)v1;
  }
  #pragma unroll
  for (int t = 0; t < 2; ++t) {
    v16h bf = fragBT_f32(w_input, TT, lm + 16*t, lhi, 0, TT);
    v8f d = {0.f,0.f,0.f,0.f,0.f,0.f,0.f,0.f};
    d = wmma32(ax, bf, d);
    #pragma unroll
    for (int v = 0; v < 8; ++v)
      s_xc[v + 8*lhi][lm + 16*t] = d[v] + b_input[lm + 16*t];
  }
  for (int k = half*16; k < half*16 + 16; ++k)
    s_xc[node][HH + k] = node_emb[(size_t)(n0 + node)*HH + k];

  // ---- num = qp @ kv ----
  v16h aq0 = fragA_f16(qp, RR, n0 + lm, lhi, 0);
  v16h aq1 = fragA_f16(qp, RR, n0 + lm, lhi, 32);
  const float* kvb = kv + (size_t)b * H2 * RR;
  #pragma unroll
  for (int t = 0; t < 4; ++t) {
    v8f d = {0.f,0.f,0.f,0.f,0.f,0.f,0.f,0.f};
    d = wmma32(aq0, fragB_f32(kvb, H2, lm + 16*t, lhi, 0),  d);
    d = wmma32(aq1, fragB_f32(kvb, H2, lm + 16*t, lhi, 32), d);
    #pragma unroll
    for (int v = 0; v < 8; ++v)
      s_h[v + 8*lhi][lm + 16*t] = d[v];
  }
  __syncthreads();

  // ---- den = qp . ksum (2 lanes per node) ----
  float dpart = 0.f;
  for (int r = half*32; r < half*32 + 32; ++r)
    dpart += (float)qp[(size_t)(n0 + node)*RR + r] * ksum[r];
  s_red[lane] = dpart; __syncthreads();
  const float den = s_red[2*node] + s_red[2*node + 1];
  __syncthreads();

  // ---- h = num/den + xc ; LayerNorm ----
  float mu = 0.f;
  for (int c = half*32; c < half*32 + 32; ++c) {
    float h = s_h[node][c] / den + s_xc[node][c];
    s_h[node][c] = h; mu += h;
  }
  s_red[lane] = mu; __syncthreads();
  mu = (s_red[2*node] + s_red[2*node + 1]) * (1.f / H2);
  __syncthreads();
  float vs = 0.f;
  for (int c = half*32; c < half*32 + 32; ++c) { float dl = s_h[node][c] - mu; vs += dl*dl; }
  s_red[lane] = vs; __syncthreads();
  const float rstd = rsqrtf((s_red[2*node] + s_red[2*node + 1]) * (1.f / H2) + 1e-5f);
  __syncthreads();
  for (int c = half*32; c < half*32 + 32; ++c)
    s_h[node][c] = (s_h[node][c] - mu) * rstd * ln_g[c] + ln_b[c];
  __syncthreads();

  // ---- out = relu([xc, h]) @ w_reg^T + b_reg  (K=128 in 4 chunks) ----
  v8f d = {0.f,0.f,0.f,0.f,0.f,0.f,0.f,0.f};
  #pragma unroll
  for (int k0 = 0; k0 < 128; k0 += 32) {
    v16h a = {};
    #pragma unroll
    for (int i = 0; i < 8; ++i) {
      int k = k0 + kidxA(i, lhi);              // pairs are even-aligned: never straddle 64
      float v0 = (k     < H2) ? s_xc[lm][k    ] : s_h[lm][k     - H2];
      float v1 = (k + 1 < H2) ? s_xc[lm][k + 1] : s_h[lm][k + 1 - H2];
      a[2*i]   = (_Float16)fmaxf(v0, 0.f);
      a[2*i+1] = (_Float16)fmaxf(v1, 0.f);
    }
    v16h bf = {};                              // cols 12..15 stay zero
    if (lm < TT) {
      #pragma unroll
      for (int i = 0; i < 8; ++i) {
        int k = k0 + kidxB(i, lhi);
        bf[2*i]   = (_Float16)w_reg[lm*128 + k];
        bf[2*i+1] = (_Float16)w_reg[lm*128 + k + 1];
      }
    }
    d = wmma32(a, bf, d);
  }
  if (lm < TT) {
    #pragma unroll
    for (int v = 0; v < 8; ++v)
      out[((size_t)b*NN + n0 + v + 8*lhi)*TT + lm] = d[v] + b_reg[lm];
  }
}

// ---------------------------------------------------------------------------
extern "C" void kernel_launch(void* const* d_in, const int* in_sizes, int n_in,
                              void* d_out, int out_size, void* d_ws, size_t ws_size,
                              hipStream_t stream) {
  (void)in_sizes; (void)n_in; (void)out_size; (void)ws_size;
  const float* x        = (const float*)d_in[0];
  const float* node_emb = (const float*)d_in[1];
  // d_in[2] source_emb_w, [3] target_emb_w, [10..13] w3/b3/w4/b4,
  // [16..17] fc1, [20..21] lnk, [25] rm2: dead branch, unused.
  const float* w_input  = (const float*)d_in[4];
  const float* b_input  = (const float*)d_in[5];
  const float* w1       = (const float*)d_in[6];
  const float* b1       = (const float*)d_in[7];
  const float* w2       = (const float*)d_in[8];
  const float* b2       = (const float*)d_in[9];
  const float* fc_w     = (const float*)d_in[14];
  const float* fc_b     = (const float*)d_in[15];
  const float* ln_g     = (const float*)d_in[18];
  const float* ln_b     = (const float*)d_in[19];
  const float* w_reg    = (const float*)d_in[22];
  const float* b_reg    = (const float*)d_in[23];
  const float* rm1      = (const float*)d_in[24];
  float* out = (float*)d_out;

  // workspace layout (~55 MB total)
  char* w = (char*)d_ws;
  size_t o = 0;
  auto alloc = [&](size_t bytes) { size_t r = o; o = (o + bytes + 255) & ~(size_t)255; return r; };
  _Float16* qp     = (_Float16*)(w + alloc(sizeof(_Float16) * (size_t)NN * RR));
  float*    ek     = (float*)   (w + alloc(sizeof(float)    * (size_t)NN * RR));
  _Float16* kpT    = (_Float16*)(w + alloc(sizeof(_Float16) * (size_t)RR * NPAD));
  _Float16* hpT    = (_Float16*)(w + alloc(sizeof(_Float16) * (size_t)BB * H2 * NPAD));
  float*    kvpart = (float*)   (w + alloc(sizeof(float)    * (size_t)BB * 16 * SEG * 256));
  float*    kvm    = (float*)   (w + alloc(sizeof(float)    * (size_t)BB * H2 * H2));
  float*    ksum   = (float*)   (w + alloc(sizeof(float)    * RR));
  float*    bmax   = (float*)   (w + alloc(sizeof(float)    * NT));
  float*    Mg     = (float*)   (w + alloc(sizeof(float)    * 4));

  k0_zero  <<<16,    256, 0, stream>>>(hpT);
  k1_node  <<<NT,     32, 0, stream>>>(node_emb, w1, b1, w2, b2, rm1, qp, ek, bmax);
  k2_max   <<<1,     256, 0, stream>>>(bmax, Mg);
  k3_kp    <<<((size_t)RR*NPAD + 255)/256, 256, 0, stream>>>(ek, Mg, kpT);
  k3b_ksum <<<RR,    256, 0, stream>>>(kpT, ksum);
  k4_xc    <<<BB*NT,  32, 0, stream>>>(x, node_emb, w_input, b_input, fc_w, fc_b, hpT);
  k5_kv    <<<BB*16*SEG, 32, 0, stream>>>(kpT, hpT, kvpart);
  k5b_red  <<<64,    256, 0, stream>>>(kvpart, kvm);
  k6_final <<<BB*NT,  32, 0, stream>>>(x, node_emb, w_input, b_input, qp, kvm, ksum,
                                       ln_g, ln_b, w_reg, b_reg, out);
}
